// GQASelfAttention_21741124452773
// MI455X (gfx1250) — matrix operand order repaired and checked by reading, not compiled, not run on hardware
//
#include <hip/hip_runtime.h>
#include <hip/hip_bf16.h>

// ---------------- types ----------------
typedef __bf16 bf16_t;
typedef __attribute__((ext_vector_type(16))) __bf16   v16bf;
typedef __attribute__((ext_vector_type(8)))  float    v8f;
typedef __attribute__((ext_vector_type(4)))  unsigned int u32x4;

#define WMMA_BF16(A_, B_, C_) \
  __builtin_amdgcn_wmma_f32_16x16x32_bf16(false, (A_), false, (B_), (short)0, (C_), false, false)

// ---------------- helpers ----------------
__device__ inline unsigned short bf_bits(float f) {
  union { float f; unsigned int u; } a; a.f = f;
  unsigned int r = a.u + 0x7FFFu + ((a.u >> 16) & 1u);
  return (unsigned short)(r >> 16);
}
__device__ inline bf16_t to_bf(float f) {
  union { unsigned short s; bf16_t h; } o; o.s = bf_bits(f); return o.h;
}
__device__ inline unsigned int pack2(float lo, float hi) {
  return (unsigned int)bf_bits(lo) | ((unsigned int)bf_bits(hi) << 16);
}
__device__ inline v8f v8f_zero() {
  v8f z;
#pragma unroll
  for (int i = 0; i < 8; ++i) z[i] = 0.0f;
  return z;
}

union FragU { u32x4 q[2]; v16bf v; };

// A-matrix fragment (16xK tile, row-major bf16 in memory, NT orientation).
// ISA layout: lane l -> row (l&15); half0 holds K = kb..kb+7 (V0-3) and
// kb+16..kb+23 (V4-7) with kb = (l>>4)*8.
__device__ inline v16bf load_a_frag(const bf16_t* tile, int ld, int lane) {
  int r  = lane & 15;
  int kb = (lane >> 4) * 8;
  const bf16_t* p = tile + (size_t)r * ld + kb;
  FragU f;
  f.q[0] = *(const u32x4*)(p);
  f.q[1] = *(const u32x4*)(p + 16);
  return f.v;
}

// B-matrix fragment (KxN) built from an NT source W[n][k] (row-major).
// ISA layout: lane l -> column (l&15); half0 holds K = kb..kb+7 (V0-3) and
// kb+16..kb+23 (V4-7) with kb = (l>>4)*8.
__device__ inline v16bf load_b_frag(const bf16_t* tile, int ld, int lane) {
  int n  = lane & 15;
  int kb = (lane >> 4) * 8;
  const bf16_t* p = tile + (size_t)n * ld + kb;
  FragU f;
  f.q[0] = *(const u32x4*)(p);
  f.q[1] = *(const u32x4*)(p + 16);
  return f.v;
}

// A fragment built on the fly from fp32 memory (convert+pack to bf16).
__device__ inline v16bf load_a_frag_f32(const float* tile, int ld, int lane) {
  int r  = lane & 15;
  int kb = (lane >> 4) * 8;
  const float* p = tile + (size_t)r * ld + kb;
  union { unsigned int w[8]; v16bf v; } f;
#pragma unroll
  for (int i = 0; i < 4; ++i) f.w[i]     = pack2(p[2 * i],      p[2 * i + 1]);
#pragma unroll
  for (int i = 0; i < 4; ++i) f.w[4 + i] = pack2(p[16 + 2 * i], p[16 + 2 * i + 1]);
  return f.v;
}

// ---------------- kernels ----------------

// fp32 -> bf16 elementwise
__global__ void cvt_bf16(const float* __restrict__ src, bf16_t* __restrict__ dst, int n) {
  int i = blockIdx.x * 256 + threadIdx.x;
  if (i < n) dst[i] = to_bf(src[i]);
}

// Generic NT GEMM: C[m][n] = sum_k A[m][k]*B[n][k] + bias[n]   (bf16 in, f32 out)
// block = 256 threads (8 waves); wave w -> rows [by*128 + w*16, +16), cols [bx*64, +64)
// Ping-pong double buffer, unrolled x2: loads for step k+32 are issued while step k's
// WMMAs run, and buffers alternate physical registers so the back-edge needs NO
// register copies (avoids the v_dual_mov_b32 rotation the single-buffer version got).
// Requires K % 64 == 0 and K >= 128 (all call sites use K = 2048).
__global__ void __launch_bounds__(256)
gemm_nt(const bf16_t* __restrict__ A, int lda,
        const bf16_t* __restrict__ Bm, int ldb,
        const float* __restrict__ bias,
        float* __restrict__ C, int ldc, int K) {
  int lane = threadIdx.x & 31;
  int wave = threadIdx.x >> 5;
  int m0 = blockIdx.y * 128 + wave * 16;
  int n0 = blockIdx.x * 64;

  v8f acc[4];
#pragma unroll
  for (int n = 0; n < 4; ++n) acc[n] = v8f_zero();

  const bf16_t* arow = A + (size_t)m0 * lda;
  const bf16_t* brow = Bm + (size_t)n0 * ldb;

  // prologue: buffer 0 holds k = 0
  v16bf a0 = load_a_frag(arow, lda, lane);
  v16bf b0[4];
#pragma unroll
  for (int n = 0; n < 4; ++n) b0[n] = load_b_frag(brow + (size_t)(n * 16) * ldb, ldb, lane);

  v16bf a1;
  v16bf b1[4];

#pragma unroll 1
  for (int k0 = 0; k0 < K - 64; k0 += 64) {
    // stage A: loads for k0+32 into buffer 1, compute on buffer 0 (k0)
    __builtin_prefetch(arow + k0 + 128, 0, 3);
    a1 = load_a_frag(arow + k0 + 32, lda, lane);
#pragma unroll
    for (int n = 0; n < 4; ++n)
      b1[n] = load_b_frag(brow + (size_t)(n * 16) * ldb + k0 + 32, ldb, lane);
#pragma unroll
    for (int n = 0; n < 4; ++n)
      acc[n] = WMMA_BF16(a0, b0[n], acc[n]);

    // stage B: loads for k0+64 into buffer 0, compute on buffer 1 (k0+32)
    a0 = load_a_frag(arow + k0 + 64, lda, lane);
#pragma unroll
    for (int n = 0; n < 4; ++n)
      b0[n] = load_b_frag(brow + (size_t)(n * 16) * ldb + k0 + 64, ldb, lane);
#pragma unroll
    for (int n = 0; n < 4; ++n)
      acc[n] = WMMA_BF16(a1, b1[n], acc[n]);
  }

  // tail: k = K-64 in buffer 0; load K-32 into buffer 1; no further loads
  a1 = load_a_frag(arow + K - 32, lda, lane);
#pragma unroll
  for (int n = 0; n < 4; ++n)
    b1[n] = load_b_frag(brow + (size_t)(n * 16) * ldb + K - 32, ldb, lane);
#pragma unroll
  for (int n = 0; n < 4; ++n)
    acc[n] = WMMA_BF16(a0, b0[n], acc[n]);
#pragma unroll
  for (int n = 0; n < 4; ++n)
    acc[n] = WMMA_BF16(a1, b1[n], acc[n]);

  int r = lane & 15, half = lane >> 4;
#pragma unroll
  for (int n = 0; n < 4; ++n) {
    int col = n0 + n * 16 + r;
    float bv = bias[col];
#pragma unroll
    for (int v = 0; v < 8; ++v) {
      int row = m0 + v + 8 * half;
      C[(size_t)row * ldc + col] = acc[n][v] + bv;
    }
  }
}

// Fused RoPE (split-half variant) + RMSNorm. One wave per (b,t,head).
// In:  X (B,T,H,128) fp32     Out: Y (B,H,T,128) bf16 (transposed for attention)
__global__ void __launch_bounds__(128)
rope_norm(const float* __restrict__ X, const float* __restrict__ cosT,
          const float* __restrict__ sinT, const float* __restrict__ w,
          bf16_t* __restrict__ Y, int nheads) {
  int lane = threadIdx.x & 31;
  int wv = blockIdx.x * 4 + (threadIdx.x >> 5);
  int h = wv % nheads;
  int t = (wv / nheads) & 2047;
  int b = wv / (nheads * 2048);

  const float2* in = (const float2*)(X + (size_t)wv * 128);
  float2 A  = in[lane];        // dims 2l, 2l+1
  float2 Bp = in[lane + 32];   // dims 2l+64, 2l+65
  float c1 = cosT[t * 64 + lane],      s1 = sinT[t * 64 + lane];
  float c2 = cosT[t * 64 + lane + 32], s2 = sinT[t * 64 + lane + 32];

  float o0 = A.x * c1 - A.y * s1;    // out dim lane
  float o1 = Bp.x * c2 - Bp.y * s2;  // out dim lane+32
  float o2 = A.x * s1 + A.y * c1;    // out dim lane+64
  float o3 = Bp.x * s2 + Bp.y * c2;  // out dim lane+96

  float ss = o0 * o0 + o1 * o1 + o2 * o2 + o3 * o3;
  ss += __shfl_xor(ss, 1, 32);
  ss += __shfl_xor(ss, 2, 32);
  ss += __shfl_xor(ss, 4, 32);
  ss += __shfl_xor(ss, 8, 32);
  ss += __shfl_xor(ss, 16, 32);
  float r = rsqrtf(ss * (1.0f / 128.0f) + 1e-6f);

  bf16_t* out = Y + (((size_t)b * nheads + h) * 2048 + t) * 128;
  out[lane]      = to_bf(o0 * r * w[lane]);
  out[lane + 32] = to_bf(o1 * r * w[lane + 32]);
  out[lane + 64] = to_bf(o2 * r * w[lane + 64]);
  out[lane + 96] = to_bf(o3 * r * w[lane + 96]);
}

// V: (B,T,HKV,128) fp32 -> (B,HKV,128,T) bf16 (d-major so PV B-frags are contiguous)
__global__ void vtrans(const float* __restrict__ Vf, bf16_t* __restrict__ Vt) {
  int i = blockIdx.x * 256 + threadIdx.x;   // over B*T*HKV*HD = 2^22
  int d = i & 127;
  int h = (i >> 7) & 3;
  int t = (i >> 9) & 2047;
  int b = i >> 20;
  Vt[(((size_t)b * 4 + h) * 128 + d) * 2048 + t] = to_bf(Vf[i]);
}

// Flash attention: one wave per (b, hq, 16-row q tile). 32 keys / iteration.
// Q: (B,16,T,128) bf16   K: (B,4,T,128) bf16   Vt: (B,4,128,T) bf16
// Out: (B,T,16,128) fp32
__global__ void __launch_bounds__(256)
attn(const bf16_t* __restrict__ Q, const bf16_t* __restrict__ Kh,
     const bf16_t* __restrict__ Vt, float* __restrict__ Out) {
  __shared__ __align__(16) bf16_t sm[8][16 * 32];   // per-wave 16x32 bf16 P tile

  int lane = threadIdx.x & 31;
  int wave = threadIdx.x >> 5;
  int gw = blockIdx.x * 8 + wave;   // 0..8191
  int qt = gw & 127;
  int hq = (gw >> 7) & 15;
  int b  = gw >> 11;
  int hk = hq >> 2;                 // REP = 4
  int qt16 = qt * 16;
  int half = lane >> 4;

  const bf16_t* qbase = Q  + (((size_t)b * 16 + hq) * 2048 + qt16) * 128;
  const bf16_t* kbase = Kh + ((size_t)b * 4 + hk) * 2048 * 128;
  const bf16_t* vbase = Vt + ((size_t)b * 4 + hk) * 128 * 2048;

  v16bf qf[4];
#pragma unroll
  for (int kk = 0; kk < 4; ++kk) qf[kk] = load_a_frag(qbase + kk * 32, 128, lane);

  v8f o[8];
#pragma unroll
  for (int n = 0; n < 8; ++n) o[n] = v8f_zero();
  float mrow[8], lrow[8];
#pragma unroll
  for (int v = 0; v < 8; ++v) { mrow[v] = -1e30f; lrow[v] = 0.0f; }

  const float scale = 0.08838834764831845f;   // 1/sqrt(128)

  for (int kb0 = 0; kb0 < qt16 + 16; kb0 += 32) {
    // ---- batch all 8 K-fragment loads (one clause / one wait), then 8 WMMAs ----
    v16bf kf[8];
#pragma unroll
    for (int kk = 0; kk < 4; ++kk) {
      kf[kk]     = load_b_frag(kbase + (size_t)(kb0)      * 128 + kk * 32, 128, lane);
      kf[4 + kk] = load_b_frag(kbase + (size_t)(kb0 + 16) * 128 + kk * 32, 128, lane);
    }
    v8f s0 = v8f_zero(), s1 = v8f_zero();
#pragma unroll
    for (int kk = 0; kk < 4; ++kk) s0 = WMMA_BF16(qf[kk], kf[kk],     s0);
#pragma unroll
    for (int kk = 0; kk < 4; ++kk) s1 = WMMA_BF16(qf[kk], kf[4 + kk], s1);

    // ---- online softmax (fp32), rows replicated per 16-lane half ----
    int c0 = kb0 + (lane & 15);
    int c1 = c0 + 16;
    float alpha[8];
#pragma unroll
    for (int v = 0; v < 8; ++v) {
      int row = qt16 + 8 * half + v;
      float a0 = s0[v] * scale; a0 = (c0 <= row) ? a0 : -1e30f;
      float a1 = s1[v] * scale; a1 = (c1 <= row) ? a1 : -1e30f;
      float rm = fmaxf(a0, a1);
      rm = fmaxf(rm, __shfl_xor(rm, 1, 32));
      rm = fmaxf(rm, __shfl_xor(rm, 2, 32));
      rm = fmaxf(rm, __shfl_xor(rm, 4, 32));
      rm = fmaxf(rm, __shfl_xor(rm, 8, 32));
      float mn = fmaxf(mrow[v], rm);
      float al = __expf(mrow[v] - mn);
      float p0 = __expf(a0 - mn);
      float p1 = __expf(a1 - mn);
      float rs = p0 + p1;
      rs += __shfl_xor(rs, 1, 32);
      rs += __shfl_xor(rs, 2, 32);
      rs += __shfl_xor(rs, 4, 32);
      rs += __shfl_xor(rs, 8, 32);
      lrow[v] = lrow[v] * al + rs;
      mrow[v] = mn;
      alpha[v] = al;
      // stash P as bf16 in the A-tile layout source (row-major 16x32)
      int sr = v + 8 * half;
      sm[wave][sr * 32 + (lane & 15)]      = to_bf(p0);
      sm[wave][sr * 32 + 16 + (lane & 15)] = to_bf(p1);
    }

    __builtin_amdgcn_wave_barrier();          // keep LDS stores before reload
    v16bf pf = load_a_frag(&sm[wave][0], 32, lane);   // DS ops are in-order per wave

    // ---- O = alpha*O + P*V (V-frags batched in groups of 4) ----
#pragma unroll
    for (int n = 0; n < 8; ++n) {
#pragma unroll
      for (int v = 0; v < 8; ++v) o[n][v] *= alpha[v];
    }
#pragma unroll
    for (int g = 0; g < 2; ++g) {
      v16bf vfr[4];
#pragma unroll
      for (int n = 0; n < 4; ++n)
        vfr[n] = load_b_frag(vbase + (size_t)((g * 4 + n) * 16) * 2048 + kb0, 2048, lane);
#pragma unroll
      for (int n = 0; n < 4; ++n)
        o[g * 4 + n] = WMMA_BF16(pf, vfr[n], o[g * 4 + n]);
    }
  }

#pragma unroll
  for (int v = 0; v < 8; ++v) lrow[v] = 1.0f / lrow[v];
#pragma unroll
  for (int n = 0; n < 8; ++n) {
#pragma unroll
    for (int v = 0; v < 8; ++v) {
      int t = qt16 + v + 8 * half;
      int d = n * 16 + (lane & 15);
      Out[(((size_t)b * 2048 + t) * 16 + hq) * 128 + d] = o[n][v] * lrow[v];
    }
  }
}

// Per-head gate: g = sigmoid(out @ Wg^T + bg); emit bf16 (out*g) in (B*T, C) layout.
// Rows m = ((b*T+t)*16 + h), K = N = 128. One wave per 16-row tile, 8 accumulators.
__global__ void __launch_bounds__(256)
gate_kernel(const float* __restrict__ Af, const bf16_t* __restrict__ Wgh,
            const float* __restrict__ bg, bf16_t* __restrict__ Gh) {
  int lane = threadIdx.x & 31;
  int wave = threadIdx.x >> 5;
  int m0 = (blockIdx.x * 8 + wave) * 16;

  v8f acc[8];
#pragma unroll
  for (int n = 0; n < 8; ++n) acc[n] = v8f_zero();

#pragma unroll
  for (int kk = 0; kk < 4; ++kk) {
    v16bf a = load_a_frag_f32(Af + (size_t)m0 * 128 + kk * 32, 128, lane);
    v16bf bfr[8];
#pragma unroll
    for (int n = 0; n < 8; ++n)
      bfr[n] = load_b_frag(Wgh + (size_t)(n * 16) * 128 + kk * 32, 128, lane);
#pragma unroll
    for (int n = 0; n < 8; ++n)
      acc[n] = WMMA_BF16(a, bfr[n], acc[n]);
  }

  int r = lane & 15, half = lane >> 4;
#pragma unroll
  for (int n = 0; n < 8; ++n) {
    int col = n * 16 + r;
    float bgc = bg[col];
#pragma unroll
    for (int v = 0; v < 8; ++v) {
      int row = m0 + v + 8 * half;                  // = (b*T+t)*16 + h
      float g  = acc[n][v] + bgc;
      float sg = 1.0f / (1.0f + __expf(-g));
      float ov = Af[(size_t)row * 128 + col];
      Gh[((size_t)(row >> 4)) * 2048 + (size_t)(row & 15) * 128 + col] = to_bf(ov * sg);
    }
  }
}

// ---------------- host orchestration ----------------
extern "C" void kernel_launch(void* const* d_in, const int* in_sizes, int n_in,
                              void* d_out, int out_size, void* d_ws, size_t ws_size,
                              hipStream_t stream) {
  (void)in_sizes; (void)n_in; (void)out_size; (void)ws_size;

  const float* x    = (const float*)d_in[0];    // (4,2048,2048)
  const float* Wq   = (const float*)d_in[1];    // (2048,2048)
  const float* bq   = (const float*)d_in[2];
  const float* Wk   = (const float*)d_in[3];    // (512,2048)
  const float* bk   = (const float*)d_in[4];
  const float* Wv   = (const float*)d_in[5];    // (512,2048)
  const float* bv   = (const float*)d_in[6];
  const float* qn_w = (const float*)d_in[7];    // (128,)
  const float* kn_w = (const float*)d_in[8];
  const float* Wg   = (const float*)d_in[9];    // (128,128)
  const float* bg   = (const float*)d_in[10];
  const float* Wo   = (const float*)d_in[11];   // (2048,2048)
  const float* bo   = (const float*)d_in[12];
  const float* cosT = (const float*)d_in[13];   // (2048,64)
  const float* sinT = (const float*)d_in[14];

  // workspace layout (256B-aligned slices)
  size_t off = 0;
  auto alloc = [&](size_t bytes) -> void* {
    void* p = (char*)d_ws + off;
    off += (bytes + 255) & ~(size_t)255;
    return p;
  };
  bf16_t* xh  = (bf16_t*)alloc(16777216ull * 2);   // x bf16
  bf16_t* Wqh = (bf16_t*)alloc(4194304ull * 2);
  bf16_t* Wkh = (bf16_t*)alloc(1048576ull * 2);
  bf16_t* Wvh = (bf16_t*)alloc(1048576ull * 2);
  bf16_t* Wgh = (bf16_t*)alloc(16384ull * 2);
  bf16_t* Woh = (bf16_t*)alloc(4194304ull * 2);
  float*  qf  = (float*)alloc(16777216ull * 4);    // raw q; reused as attn out
  float*  kf  = (float*)alloc(4194304ull * 4);
  float*  vf  = (float*)alloc(4194304ull * 4);
  bf16_t* qh  = (bf16_t*)alloc(16777216ull * 2);   // (B,16,T,128)
  bf16_t* kh  = (bf16_t*)alloc(4194304ull * 2);    // (B,4,T,128)
  bf16_t* vh  = (bf16_t*)alloc(4194304ull * 2);    // (B,4,128,T)
  bf16_t* gh  = (bf16_t*)alloc(16777216ull * 2);   // gated out bf16, (B*T, C)
  (void)alloc(65536);                              // slack for bounded tail over-reads
  float* attnf = qf;                               // alias: qf dead after rope_norm

  // 1) fp32 -> bf16 conversions
  cvt_bf16<<<(16777216 + 255) / 256, 256, 0, stream>>>(x,  xh,  16777216);
  cvt_bf16<<<(4194304  + 255) / 256, 256, 0, stream>>>(Wq, Wqh, 4194304);
  cvt_bf16<<<(1048576  + 255) / 256, 256, 0, stream>>>(Wk, Wkh, 1048576);
  cvt_bf16<<<(1048576  + 255) / 256, 256, 0, stream>>>(Wv, Wvh, 1048576);
  cvt_bf16<<<(16384    + 255) / 256, 256, 0, stream>>>(Wg, Wgh, 16384);
  cvt_bf16<<<(4194304  + 255) / 256, 256, 0, stream>>>(Wo, Woh, 4194304);

  // 2) QKV projections (NT GEMMs, M = B*T = 8192)
  gemm_nt<<<dim3(32, 64), 256, 0, stream>>>(xh, 2048, Wqh, 2048, bq, qf, 2048, 2048);
  gemm_nt<<<dim3(8,  64), 256, 0, stream>>>(xh, 2048, Wkh, 2048, bk, kf, 512,  2048);
  gemm_nt<<<dim3(8,  64), 256, 0, stream>>>(xh, 2048, Wvh, 2048, bv, vf, 512,  2048);

  // 3) RoPE + RMSNorm (+ transpose to head-major), V transpose
  rope_norm<<<32768, 128, 0, stream>>>(qf, cosT, sinT, qn_w, qh, 16);
  rope_norm<<<8192,  128, 0, stream>>>(kf, cosT, sinT, kn_w, kh, 4);
  vtrans<<<16384, 256, 0, stream>>>(vf, vh);

  // 4) flash attention (8192 wave-tiles, 8 waves/block)
  attn<<<1024, 256, 0, stream>>>(qh, kh, vh, attnf);

  // 5) per-head sigmoid gate
  gate_kernel<<<1024, 256, 0, stream>>>(attnf, Wgh, bg, gh);

  // 6) output projection
  gemm_nt<<<dim3(32, 64), 256, 0, stream>>>(gh, 2048, Woh, 2048, bo,
                                            (float*)d_out, 2048, 2048);
}
